// AttentionBlock_33612414059172
// MI455X (gfx1250) — matrix operand order repaired and checked by reading, not compiled, not run on hardware
//
#include <hip/hip_runtime.h>

// ---------------------------------------------------------------------------
// AttentionBlock for MI455X (gfx1250, wave32, WMMA).
// B=4, C=512, T=2048, H=8, dh=64, groups=32.
// Pipeline: f32->f16 weight cvt | GroupNorm->h_t[B,T,C] f16 |
//           QKV WMMA GEMM (32x64/wave) -> q,k [B,H,T,dh] (scaled), v_t[B,H,dh,T] |
//           flash-attn, 64-col steps (WMMA QK^T + online softmax + WMMA PV) |
//           proj WMMA GEMM (32x64/wave) + bias + residual -> out[B,C,T] f32.
// All intermediates f16 in workspace (~42 MB, L2-resident on 192MB L2).
// ---------------------------------------------------------------------------

#define B_SZ  4
#define C_CH  512
#define T_SEQ 2048
#define H_NUM 8
#define DH    64
#define QK_SCALE 0.35355339059327373f   // 1/sqrt(sqrt(64))

typedef __attribute__((ext_vector_type(8)))  _Float16 v8h;
typedef __attribute__((ext_vector_type(16))) _Float16 v16h;
typedef __attribute__((ext_vector_type(8)))  float    v8f;

// Build a 16-half fragment from two contiguous 16B chunks (2x b128 loads).
__device__ __forceinline__ v16h load_pair(const _Float16* p0, const _Float16* p1) {
  v8h a = *(const v8h*)p0;
  v8h b = *(const v8h*)p1;
  v16h r;
#pragma unroll
  for (int i = 0; i < 8; ++i) { r[i] = a[i]; r[i + 8] = b[i]; }
  return r;
}

__device__ __forceinline__ v8f wmma_f16(v16h a, v16h b, v8f c) {
  return __builtin_amdgcn_wmma_f32_16x16x32_f16(false, a, false, b, (short)0, c,
                                                false, false);
}

// ---------------------------------------------------------------------------
// f32 -> f16 elementwise (weights)
// ---------------------------------------------------------------------------
__global__ __launch_bounds__(256) void cvt_kernel(const float* __restrict__ src,
                                                  _Float16* __restrict__ dst, int n) {
  int i = blockIdx.x * 256 + threadIdx.x;
  if (i < n) dst[i] = (_Float16)src[i];
}

// ---------------------------------------------------------------------------
// GroupNorm: x[B,C,T] f32 -> h_t[B,T,C] f16 (token-major for WMMA B-fragments)
// One block per (b, group); a group's 16 channels are contiguous in x.
// ---------------------------------------------------------------------------
__global__ __launch_bounds__(256) void gn_kernel(const float* __restrict__ x,
                                                 const float* __restrict__ gamma,
                                                 const float* __restrict__ beta,
                                                 _Float16* __restrict__ h_t) {
  const int b   = blockIdx.x >> 5;     // 32 groups
  const int grp = blockIdx.x & 31;
  const int c0  = grp * 16;
  const float* xb = x + ((size_t)b * C_CH + c0) * T_SEQ;  // flat 16*T block

  float s = 0.f, s2 = 0.f;
  for (int i = threadIdx.x; i < 16 * T_SEQ; i += 256) {
    float v = xb[i];
    s += v; s2 += v * v;
  }
#pragma unroll
  for (int m = 1; m < 32; m <<= 1) {
    s  += __shfl_xor(s,  m);
    s2 += __shfl_xor(s2, m);
  }
  __shared__ float red[2][8];
  const int wave = threadIdx.x >> 5;
  if ((threadIdx.x & 31) == 0) { red[0][wave] = s; red[1][wave] = s2; }
  __syncthreads();
  float ts = 0.f, ts2 = 0.f;
#pragma unroll
  for (int w = 0; w < 8; ++w) { ts += red[0][w]; ts2 += red[1][w]; }
  const float inv_n = 1.0f / (16.0f * T_SEQ);
  const float mean  = ts * inv_n;
  const float var   = ts2 * inv_n - mean * mean;
  const float rstd  = rsqrtf(var + 1e-5f);

  for (int i = threadIdx.x; i < 16 * T_SEQ; i += 256) {
    const int cl = i >> 11;            // T_SEQ == 2048
    const int t  = i & (T_SEQ - 1);
    const int c  = c0 + cl;
    const float v = (xb[i] - mean) * rstd * gamma[c] + beta[c];
    h_t[((size_t)b * T_SEQ + t) * C_CH + c] = (_Float16)v;
  }
}

// ---------------------------------------------------------------------------
// QKV GEMM: qkv[o,t] = sum_c Wqkv[o,c] * h[c,t] + b[o], o in [0,3C)
// A = Wqkv[3C,C] (f16, row-major), B = h_t[T,C] (token-major => contiguous
// B-fragments). 4 waves/block; wave owns 32(M) x 64(N): 2 M-subtiles share
// each B-fragment -> 8 WMMA / 12 b128 loads per K-step of 32.
// ---------------------------------------------------------------------------
__global__ __launch_bounds__(128) void qkv_gemm(const _Float16* __restrict__ w16,
                                                const _Float16* __restrict__ h_t,
                                                const float* __restrict__ bias,
                                                _Float16* __restrict__ qb,
                                                _Float16* __restrict__ kb,
                                                _Float16* __restrict__ vtb) {
  const int wave = threadIdx.x >> 5;
  const int lane = threadIdx.x & 31;
  const int g    = lane >> 4;
  const int ln   = lane & 15;
  const int n0   = blockIdx.x * 64;
  const int m0   = blockIdx.y * 128 + wave * 32;
  const int b    = blockIdx.z;
  const _Float16* hb = h_t + (size_t)b * T_SEQ * C_CH;

  v8f acc[2][4];
#pragma unroll
  for (int mt = 0; mt < 2; ++mt)
#pragma unroll
    for (int nb = 0; nb < 4; ++nb) acc[mt][nb] = (v8f){0,0,0,0,0,0,0,0};

  const _Float16* arow0 = w16 + (size_t)(m0 + ln) * C_CH;
  const _Float16* arow1 = arow0 + (size_t)16 * C_CH;

  for (int k0 = 0; k0 < C_CH; k0 += 32) {
    v16h af0 = load_pair(arow0 + k0 + 8 * g, arow0 + k0 + 16 + 8 * g);
    v16h af1 = load_pair(arow1 + k0 + 8 * g, arow1 + k0 + 16 + 8 * g);
#pragma unroll
    for (int nb = 0; nb < 4; ++nb) {
      const _Float16* brow = hb + (size_t)(n0 + nb * 16 + ln) * C_CH + k0 + 16 * g;
      v16h bf = load_pair(brow - 8 * g, brow - 8 * g + 16);
      acc[0][nb] = wmma_f16(af0, bf, acc[0][nb]);
      acc[1][nb] = wmma_f16(af1, bf, acc[1][nb]);
    }
  }

#pragma unroll
  for (int mt = 0; mt < 2; ++mt) {
#pragma unroll
    for (int nb = 0; nb < 4; ++nb) {
#pragma unroll
      for (int r = 0; r < 8; ++r) {
        const int o = m0 + mt * 16 + r + 8 * g;
        const int t = n0 + nb * 16 + ln;
        float val = acc[mt][nb][r] + bias[o];
        const int head = o / 192;
        const int rem  = o % 192;
        const int sel  = rem / DH;
        const int d    = rem % DH;
        const size_t bh = (size_t)b * H_NUM + head;
        if (sel == 0)      qb[(bh * T_SEQ + t) * DH + d] = (_Float16)(val * QK_SCALE);
        else if (sel == 1) kb[(bh * T_SEQ + t) * DH + d] = (_Float16)(val * QK_SCALE);
        else               vtb[(bh * DH + d) * T_SEQ + t] = (_Float16)val;
      }
    }
  }
}

// ---------------------------------------------------------------------------
// Flash attention: one wave per 16-token row tile of one (b,h).
// 64-column steps: 8 WMMA for S (16x64), one online-softmax update
// (shuffle reductions + acc rescale amortized over 64 cols), P relayout
// C->A via per-wave LDS tile, 8 WMMA for PV.
// ---------------------------------------------------------------------------
__global__ __launch_bounds__(128) void attn_kernel(const _Float16* __restrict__ q,
                                                   const _Float16* __restrict__ k,
                                                   const _Float16* __restrict__ vt,
                                                   _Float16* __restrict__ attn) {
  __shared__ __align__(16) _Float16 p_lds_all[4][16 * 64];
  const int wave = threadIdx.x >> 5;
  const int lane = threadIdx.x & 31;
  const int g    = lane >> 4;
  const int ln   = lane & 15;
  _Float16* p_lds = p_lds_all[wave];

  const int tile = blockIdx.x * 4 + wave;       // [0, B*H*T/16)
  const int t0   = (tile & 127) * 16;           // T/16 == 128
  const int bh   = tile >> 7;                   // b*H + head
  const _Float16* qbp = q  + (size_t)bh * T_SEQ * DH;
  const _Float16* kbp = k  + (size_t)bh * T_SEQ * DH;
  const _Float16* vbp = vt + (size_t)bh * DH * T_SEQ;

  // Preload Q A-fragments (two K-chunks covering dh=64)
  const _Float16* qrow = qbp + (size_t)(t0 + ln) * DH;
  v16h qa0 = load_pair(qrow +  0 + 8 * g, qrow + 16 + 8 * g);
  v16h qa1 = load_pair(qrow + 32 + 8 * g, qrow + 48 + 8 * g);

  v8f acc[4];
#pragma unroll
  for (int nb = 0; nb < 4; ++nb) acc[nb] = (v8f){0,0,0,0,0,0,0,0};
  float m_run[8], l_run[8];
#pragma unroll
  for (int r = 0; r < 8; ++r) { m_run[r] = -3.0e38f; l_run[r] = 0.0f; }

  for (int s0 = 0; s0 < T_SEQ; s0 += 64) {
    // ---- scores: 16x64 block as four 16x16 C-fragments ----
    v8f sf[4];
#pragma unroll
    for (int sb = 0; sb < 4; ++sb) {
      v8f z = (v8f){0,0,0,0,0,0,0,0};
      const _Float16* krow = kbp + (size_t)(s0 + sb * 16 + ln) * DH;
      v16h kf0 = load_pair(krow + 8 * g,      krow + 16 + 8 * g);
      v16h kf1 = load_pair(krow + 32 + 8 * g, krow + 48 + 8 * g);
      z = wmma_f16(qa0, kf0, z);
      z = wmma_f16(qa1, kf1, z);
      sf[sb] = z;
    }

    // ---- online softmax, one update per 64 columns ----
    float alpha[8];
#pragma unroll
    for (int r = 0; r < 8; ++r) {
      float mx = fmaxf(fmaxf(sf[0][r], sf[1][r]), fmaxf(sf[2][r], sf[3][r]));
#pragma unroll
      for (int m = 1; m < 16; m <<= 1) mx = fmaxf(mx, __shfl_xor(mx, m));
      const float mn = fmaxf(m_run[r], mx);
      const float al = __expf(m_run[r] - mn);
      float ps = 0.f;
#pragma unroll
      for (int sb = 0; sb < 4; ++sb) {
        const float p = __expf(sf[sb][r] - mn);
        sf[sb][r] = p;
        ps += p;
      }
#pragma unroll
      for (int m = 1; m < 16; m <<= 1) ps += __shfl_xor(ps, m);
      l_run[r] = l_run[r] * al + ps;
      m_run[r] = mn;
      alpha[r] = al;
    }
#pragma unroll
    for (int nb = 0; nb < 4; ++nb)
#pragma unroll
      for (int r = 0; r < 8; ++r) acc[nb][r] *= alpha[r];

    // ---- P: C-layout f32 -> A-layout f16 via LDS (16x64 tile) ----
#pragma unroll
    for (int r = 0; r < 8; ++r)
#pragma unroll
      for (int sb = 0; sb < 4; ++sb)
        p_lds[(r + 8 * g) * 64 + sb * 16 + ln] = (_Float16)sf[sb][r];
    asm volatile("s_wait_dscnt 0" ::: "memory");  // same-wave LDS RAW
    const _Float16* prow = p_lds + ln * 64;
    v16h pa0 = load_pair(prow +  0 + 8 * g, prow + 16 + 8 * g);
    v16h pa1 = load_pair(prow + 32 + 8 * g, prow + 48 + 8 * g);

    // ---- PV: out[16x64] += P[16x64] x V[64x64] ----
#pragma unroll
    for (int nb = 0; nb < 4; ++nb) {
      const _Float16* vcol = vbp + (size_t)(nb * 16 + ln) * T_SEQ + s0;
      v16h vf0 = load_pair(vcol + 8 * g,      vcol + 16 + 8 * g);
      v16h vf1 = load_pair(vcol + 32 + 8 * g, vcol + 48 + 8 * g);
      acc[nb] = wmma_f16(pa0, vf0, acc[nb]);
      acc[nb] = wmma_f16(pa1, vf1, acc[nb]);
    }
  }

  // ---- epilogue: normalize, write attn[B,T,C] f16 (token-major) ----
  const int b  = bh >> 3;
  const int hc = (bh & 7) * DH;
#pragma unroll
  for (int nb = 0; nb < 4; ++nb) {
#pragma unroll
    for (int r = 0; r < 8; ++r) {
      const float o  = acc[nb][r] / l_run[r];
      const int trow = t0 + r + 8 * g;
      attn[((size_t)b * T_SEQ + trow) * C_CH + hc + nb * 16 + ln] = (_Float16)o;
    }
  }
}

// ---------------------------------------------------------------------------
// Proj GEMM + bias + residual: out[b,o,t] = x[b,o,t] + sum_c Wp[o,c]*attn[c,t] + bp[o]
// Same 32x64-per-wave structure as qkv_gemm.
// ---------------------------------------------------------------------------
__global__ __launch_bounds__(128) void proj_gemm(const _Float16* __restrict__ w16,
                                                 const _Float16* __restrict__ attn,
                                                 const float* __restrict__ x,
                                                 const float* __restrict__ bias,
                                                 float* __restrict__ out) {
  const int wave = threadIdx.x >> 5;
  const int lane = threadIdx.x & 31;
  const int g    = lane >> 4;
  const int ln   = lane & 15;
  const int n0   = blockIdx.x * 64;
  const int m0   = blockIdx.y * 128 + wave * 32;
  const int b    = blockIdx.z;
  const _Float16* ab = attn + (size_t)b * T_SEQ * C_CH;

  v8f acc[2][4];
#pragma unroll
  for (int mt = 0; mt < 2; ++mt)
#pragma unroll
    for (int nb = 0; nb < 4; ++nb) acc[mt][nb] = (v8f){0,0,0,0,0,0,0,0};

  const _Float16* arow0 = w16 + (size_t)(m0 + ln) * C_CH;
  const _Float16* arow1 = arow0 + (size_t)16 * C_CH;

  for (int k0 = 0; k0 < C_CH; k0 += 32) {
    v16h af0 = load_pair(arow0 + k0 + 8 * g, arow0 + k0 + 16 + 8 * g);
    v16h af1 = load_pair(arow1 + k0 + 8 * g, arow1 + k0 + 16 + 8 * g);
#pragma unroll
    for (int nb = 0; nb < 4; ++nb) {
      const _Float16* brow = ab + (size_t)(n0 + nb * 16 + ln) * C_CH + k0 + 16 * g;
      v16h bf = load_pair(brow - 8 * g, brow - 8 * g + 16);
      acc[0][nb] = wmma_f16(af0, bf, acc[0][nb]);
      acc[1][nb] = wmma_f16(af1, bf, acc[1][nb]);
    }
  }

#pragma unroll
  for (int mt = 0; mt < 2; ++mt) {
#pragma unroll
    for (int nb = 0; nb < 4; ++nb) {
#pragma unroll
      for (int r = 0; r < 8; ++r) {
        const int o = m0 + mt * 16 + r + 8 * g;
        const int t = n0 + nb * 16 + ln;
        const size_t idx = ((size_t)b * C_CH + o) * T_SEQ + t;
        out[idx] = x[idx] + acc[mt][nb][r] + bias[o];
      }
    }
  }
}

// ---------------------------------------------------------------------------
extern "C" void kernel_launch(void* const* d_in, const int* in_sizes, int n_in,
                              void* d_out, int out_size, void* d_ws, size_t ws_size,
                              hipStream_t stream) {
  const float* x      = (const float*)d_in[0];
  const float* gamma  = (const float*)d_in[1];
  const float* beta   = (const float*)d_in[2];
  const float* w_qkv  = (const float*)d_in[3];
  const float* b_qkv  = (const float*)d_in[4];
  const float* w_proj = (const float*)d_in[5];
  const float* b_proj = (const float*)d_in[6];
  float* out = (float*)d_out;

  const size_t NTC = (size_t)B_SZ * T_SEQ * C_CH;   // 4,194,304 elems
  _Float16* ws   = (_Float16*)d_ws;                 // needs ~42 MB f16 scratch
  _Float16* h_t  = ws;                              // [B,T,C]
  _Float16* qbuf = h_t  + NTC;                      // [B,H,T,dh]
  _Float16* kbuf = qbuf + NTC;                      // [B,H,T,dh]
  _Float16* vbuf = kbuf + NTC;                      // [B,H,dh,T]
  _Float16* abuf = vbuf + NTC;                      // [B,T,C]
  _Float16* wq16 = abuf + NTC;                      // [3C,C]
  _Float16* wp16 = wq16 + (size_t)3 * C_CH * C_CH;  // [C,C]

  const int nwq = 3 * C_CH * C_CH;
  const int nwp = C_CH * C_CH;
  cvt_kernel<<<(nwq + 255) / 256, 256, 0, stream>>>(w_qkv, wq16, nwq);
  cvt_kernel<<<(nwp + 255) / 256, 256, 0, stream>>>(w_proj, wp16, nwp);
  gn_kernel<<<B_SZ * 32, 256, 0, stream>>>(x, gamma, beta, h_t);
  qkv_gemm<<<dim3(T_SEQ / 64, (3 * C_CH) / 128, B_SZ), 128, 0, stream>>>(
      wq16, h_t, b_qkv, qbuf, kbuf, vbuf);
  attn_kernel<<<(B_SZ * H_NUM * (T_SEQ / 16)) / 4, 128, 0, stream>>>(
      qbuf, kbuf, vbuf, abuf);
  proj_gemm<<<dim3(T_SEQ / 64, C_CH / 128, B_SZ), 128, 0, stream>>>(
      wp16, abuf, x, b_proj, out);
}
